// SOL_63436666962427
// MI455X (gfx1250) — compile-verified, matrix operand order checked by reading, not run on hardware
//
#include <hip/hip_runtime.h>

// ---------------------------------------------------------------------------
// loss = 1 - mean_c[(B - P_c - S0_c + 2*S1_c)/B]
//   psi[b,i] = (1/N) sum_n  2 / prod_j (1 + E_j * R_i),  E_k = exp2(C*(y_bk - tau_nk))
//   S0_c = sum_b psi[b,c], S1_c = sum_b y_true[b,c]*psi[b,c], P_c = sum_b y_true[b,c]
// ---------------------------------------------------------------------------

typedef __attribute__((ext_vector_type(16))) _Float16 v16h;
typedef __attribute__((ext_vector_type(8)))  float    v8f;

#define MCLS 16

// Kernel 1: compute psi (B x 16). Block = 256 threads = 16 b-rows x 16 n-slices.
__global__ __launch_bounds__(256) void SOL_psi_kernel(
    const float* __restrict__ y_pred, const float* __restrict__ taus,
    float* __restrict__ psi, int N) {
  __shared__ float lds[256 * 17];  // stride-17 padding: conflict-free banks

  const float C = 10.0f * 1.4426950408889634f;  // LAM * log2(e)
  const int t  = threadIdx.x;
  const int bl = t >> 4;        // local b row (0..15)
  const int s  = t & 15;        // n-slice (0..15)
  const int b  = blockIdx.x * 16 + bl;

  // Stage taus into LDS, pre-scaled by C. N == 256 == blockDim.x rows.
  for (int r = t; r < N; r += 256) {
#pragma unroll
    for (int k = 0; k < MCLS; ++k) lds[r * 17 + k] = taus[r * MCLS + k] * C;
  }
  __syncthreads();

  float yc[MCLS];
#pragma unroll
  for (int k = 0; k < MCLS; ++k) yc[k] = y_pred[b * MCLS + k] * C;

  float acc[MCLS];
#pragma unroll
  for (int i = 0; i < MCLS; ++i) acc[i] = 0.0f;

  const int NS = N >> 4;  // n-values per slice (16)
  for (int nn = 0; nn < NS; ++nn) {
    const int n = nn * 16 + s;  // lanes with distinct s -> distinct banks (pad 17)
    float E[MCLS], R[MCLS];
#pragma unroll
    for (int k = 0; k < MCLS; ++k) {
      float e = __builtin_amdgcn_exp2f(yc[k] - lds[n * 17 + k]);  // v_exp_f32
      E[k] = e;
      R[k] = __builtin_amdgcn_rcpf(e);                             // v_rcp_f32
    }
#pragma unroll
    for (int i = 0; i < MCLS; ++i) {
      float p = 1.0f;
#pragma unroll
      for (int j = 0; j < MCLS; ++j)
        p *= __builtin_fmaf(E[j], R[i], 1.0f);   // fma + mul core
      // diagonal term contributed (1 + E_i*R_i) ~= 2  ->  p_i = 2/prod
      acc[i] += 2.0f * __builtin_amdgcn_rcpf(p); // inf -> 0, matches true psi~0
    }
  }

  // Reduce the 16 n-slices per b-row through LDS (reuse tau buffer).
  __syncthreads();
#pragma unroll
  for (int i = 0; i < MCLS; ++i) lds[t * 17 + i] = acc[i];
  __syncthreads();

  // thread t -> (row bl, class s): sum over the 16 slices
  float sum = 0.0f;
#pragma unroll
  for (int ss = 0; ss < 16; ++ss) sum += lds[(bl * 16 + ss) * 17 + s];
  psi[(blockIdx.x * 16 + bl) * MCLS + s] = sum / (float)N;
}

// Kernel 2: WMMA column-sum reduction over B + final scalar loss.
// One block, 8 waves. Each wave reduces B/8 rows in chunks of 32 via
// v_wmma_f32_16x16x32_f16 with an all-ones A matrix (D rows = column sums).
// psi is split hi/lo in f16 (compensated) for near-f32 accuracy.
__global__ __launch_bounds__(256) void SOL_reduce_kernel(
    const float* __restrict__ psi, const float* __restrict__ y_true,
    float* __restrict__ out, int B) {
  __shared__ float red[3][8][16];

  const int t    = threadIdx.x;
  const int wave = t >> 5;
  const int lane = t & 31;
  const int col  = lane & 15;
  const int hi   = lane >> 4;

  v16h ones;
#pragma unroll
  for (int i = 0; i < 16; ++i) ones[i] = (_Float16)1.0f;

  v8f s0 = {};  // sum psi
  v8f s1 = {};  // sum y_true*psi
  v8f sp = {};  // sum y_true

  const int rowsPerWave = B >> 3;
  const int chunks = rowsPerWave >> 5;
  const int base0 = wave * rowsPerWave;

  for (int ch = 0; ch < chunks; ++ch) {
    // lanes 0-15 cover rows base..base+15 (col = lane), lanes 16-31 the next 16;
    // with all-ones A only the column<->lane%16 mapping matters.
    const int base = base0 + ch * 32 + hi * 16;
    v16h bp_h, bp_l, by_h, by_l, byt;
#pragma unroll
    for (int h = 0; h < 16; ++h) {
      const int r = base + h;
      float pv = psi[r * MCLS + col];
      float yv = y_true[r * MCLS + col];
      _Float16 ph = (_Float16)pv;
      bp_h[h] = ph;
      bp_l[h] = (_Float16)(pv - (float)ph);   // compensated residual
      float yp = yv * pv;
      _Float16 yh = (_Float16)yp;
      by_h[h] = yh;
      by_l[h] = (_Float16)(yp - (float)yh);
      byt[h]  = (_Float16)yv;                 // 0/1 exact in f16
    }
    s0 = __builtin_amdgcn_wmma_f32_16x16x32_f16(false, ones, false, bp_h, (short)0, s0, false, false);
    s0 = __builtin_amdgcn_wmma_f32_16x16x32_f16(false, ones, false, bp_l, (short)0, s0, false, false);
    s1 = __builtin_amdgcn_wmma_f32_16x16x32_f16(false, ones, false, by_h, (short)0, s1, false, false);
    s1 = __builtin_amdgcn_wmma_f32_16x16x32_f16(false, ones, false, by_l, (short)0, s1, false, false);
    sp = __builtin_amdgcn_wmma_f32_16x16x32_f16(false, ones, false, byt, (short)0, sp, false, false);
  }

  // D layout: VGPR0, any lane -> column (lane&15) sum (all D rows identical).
  if (lane < 16) {
    red[0][wave][col] = s0[0];
    red[1][wave][col] = s1[0];
    red[2][wave][col] = sp[0];
  }
  __syncthreads();

  if (t == 0) {
    const float invB = 1.0f / (float)B;
    float accsum = 0.0f;
    for (int c = 0; c < 16; ++c) {
      float S0 = 0.0f, S1 = 0.0f, P = 0.0f;
      for (int w = 0; w < 8; ++w) {
        S0 += red[0][w][c];
        S1 += red[1][w][c];
        P  += red[2][w][c];
      }
      // acc_c = (TP + TN) / B = (B - P - S0 + 2*S1) / B
      accsum += ((float)B - P - S0 + 2.0f * S1) * invB;
    }
    out[0] = 1.0f - accsum * (1.0f / 16.0f);
  }
}

extern "C" void kernel_launch(void* const* d_in, const int* in_sizes, int n_in,
                              void* d_out, int out_size, void* d_ws, size_t ws_size,
                              hipStream_t stream) {
  const float* y_pred = (const float*)d_in[0];  // (B,16) f32
  const float* y_true = (const float*)d_in[1];  // (B,16) f32 one-hot
  const float* taus   = (const float*)d_in[2];  // (N,16) f32

  const int B = in_sizes[0] / MCLS;  // 2048
  const int N = in_sizes[2] / MCLS;  // 256

  float* psi = (float*)d_ws;         // B*16 floats = 128 KB scratch

  SOL_psi_kernel<<<B / 16, 256, 0, stream>>>(y_pred, taus, psi, N);
  SOL_reduce_kernel<<<1, 256, 0, stream>>>(psi, y_true, (float*)d_out, B);
}
